// Attention_11184094839394
// MI455X (gfx1250) — matrix operand order repaired and checked by reading, not compiled, not run on hardware
//
#include <hip/hip_runtime.h>
#include <hip/hip_bf16.h>
#include <stdint.h>

// ---------------------------------------------------------------------------
// MI455X (gfx1250) multi-head attention: all GEMMs via v_wmma_f32_16x16x32_bf16
// Round 2: pre-cast X to bf16 so every GEMM mainloop is pure b128-load + WMMA.
// ---------------------------------------------------------------------------

typedef __attribute__((ext_vector_type(16))) __bf16 v16bf;
typedef __attribute__((ext_vector_type(8)))  float  v8f;

#define HIDDEN   1024
#define HEADS    16
#define HEAD_DIM 64
#define BATCH    8
#define SEQ      1024
#define M_TOTAL  (BATCH * SEQ)   // 8192 rows
#define NPROJ    (3 * HIDDEN)    // 3072
#define ATT_SCALE 0.125f         // 1/sqrt(64)

union AFrag {
    v16bf v;
    uint4 q[2];
    unsigned short h[16];
};

__device__ __forceinline__ unsigned short f32_to_bf16(float f) {
    union { float f; unsigned u; } x; x.f = f;
    unsigned u = x.u;
    u += 0x7FFFu + ((u >> 16) & 1u);   // round-to-nearest-even
    return (unsigned short)(u >> 16);
}

__device__ __forceinline__ float bf16_to_f32(unsigned short h) {
    union { unsigned u; float f; } x; x.u = ((unsigned)h) << 16;
    return x.f;
}

__device__ __forceinline__ v8f wmma_bf16(const AFrag& a, const AFrag& b, v8f c) {
    // (neg_a, A, neg_b, B, c_mod, C, reuse_a, reuse_b)
    return __builtin_amdgcn_wmma_f32_16x16x32_bf16(false, a.v, false, b.v,
                                                   (short)0, c, false, false);
}

__device__ __forceinline__ v8f v8f_zero() {
    v8f z = {0.f, 0.f, 0.f, 0.f, 0.f, 0.f, 0.f, 0.f};
    return z;
}

// ---------------------------------------------------------------------------
// Kernel 0: fp32 -> bf16 cast (x and weights), 4 elements / thread
// ---------------------------------------------------------------------------
__global__ void cast_f32_bf16_kernel(const float* __restrict__ src,
                                     unsigned short* __restrict__ dst, int n4) {
    int i = blockIdx.x * blockDim.x + threadIdx.x;
    if (i >= n4) return;
    float4 f = ((const float4*)src)[i];
    ushort4 o;
    o.x = f32_to_bf16(f.x);
    o.y = f32_to_bf16(f.y);
    o.z = f32_to_bf16(f.z);
    o.w = f32_to_bf16(f.w);
    ((ushort4*)dst)[i] = o;
}

// ---------------------------------------------------------------------------
// Kernel 1: in_proj  qkv = x @ w_in^T + b_in, scattered to Q/K/V^T (bf16)
//   Q,K : [B,H,N,D] row-major (d contiguous)
//   Vt  : [B,H,D,N] row-major (n contiguous)
// Block: 256 thr = 8 waves in 2(M)x4(N); wave tile 64x64; block tile 128x256.
// ---------------------------------------------------------------------------
__global__ __launch_bounds__(256) void in_proj_kernel(
    const unsigned short* __restrict__ X,   // [8192,1024] bf16
    const unsigned short* __restrict__ W,   // [3072,1024] bf16
    const float* __restrict__ bias,         // [3072]
    unsigned short* __restrict__ Qb,
    unsigned short* __restrict__ Kb,
    unsigned short* __restrict__ Vt) {
    const int lane = threadIdx.x & 31;
    const int wave = threadIdx.x >> 5;
    const int g = lane >> 4;      // lane group (0/1)
    const int lr = lane & 15;     // row/col within 16
    const int m0 = blockIdx.x * 128 + (wave & 1) * 64;
    const int n0 = blockIdx.y * 256 + (wave >> 1) * 64;

    v8f acc[4][4];
#pragma unroll
    for (int i = 0; i < 4; ++i)
#pragma unroll
        for (int j = 0; j < 4; ++j) acc[i][j] = v8f_zero();

    for (int k0 = 0; k0 < HIDDEN; k0 += 32) {
        AFrag a[4];
#pragma unroll
        for (int i = 0; i < 4; ++i) {
            const unsigned short* ap = X + (size_t)(m0 + i * 16 + lr) * HIDDEN + k0;
            a[i].q[0] = *(const uint4*)(ap + g * 16);
            a[i].q[1] = *(const uint4*)(ap + g * 16 + 8);
        }
        AFrag b[4];
#pragma unroll
        for (int j = 0; j < 4; ++j) {
            const unsigned short* bp =
                W + (size_t)(n0 + j * 16 + lr) * HIDDEN + k0 + g * 16;
            b[j].q[0] = *(const uint4*)(bp);
            b[j].q[1] = *(const uint4*)(bp + 8);
        }
#pragma unroll
        for (int i = 0; i < 4; ++i)
#pragma unroll
            for (int j = 0; j < 4; ++j)
                acc[i][j] = wmma_bf16(a[i], b[j], acc[i][j]);
    }

    // Epilogue: bias + scatter to Q / K / V^T
#pragma unroll
    for (int i = 0; i < 4; ++i) {
#pragma unroll
        for (int j = 0; j < 4; ++j) {
            const int o = n0 + j * 16 + lr;       // output feature in [0,3072)
            const float bv = bias[o];
            const int t = o >> 10;                // 0=q 1=k 2=v
            const int rem = o & 1023;
            const int h = rem >> 6;
            const int d = rem & 63;
#pragma unroll
            for (int e = 0; e < 8; ++e) {
                const int m = m0 + i * 16 + 8 * g + e;
                const int bb = m >> 10;
                const int n = m & 1023;
                const unsigned short hv = f32_to_bf16(acc[i][j][e] + bv);
                if (t == 0)
                    Qb[(((size_t)(bb * HEADS + h)) * SEQ + n) * HEAD_DIM + d] = hv;
                else if (t == 1)
                    Kb[(((size_t)(bb * HEADS + h)) * SEQ + n) * HEAD_DIM + d] = hv;
                else
                    Vt[(((size_t)(bb * HEADS + h)) * HEAD_DIM + d) * SEQ + n] = hv;
            }
        }
    }
}

// ---------------------------------------------------------------------------
// Kernel 2: attention. One wave per (16-query tile, head).
//   grid = (SEQ/16 = 64, B*H = 128), block = 32.
//   Pass1: S = scale * Q K^T -> bf16 in LDS, track row max.
//   Pass2: row sums of exp(S - max).
//   Pass3: O = exp(S - max) @ V via WMMA, divide by sum, store bf16.
// ---------------------------------------------------------------------------
__global__ __launch_bounds__(32) void attention_kernel(
    const unsigned short* __restrict__ Qb,
    const unsigned short* __restrict__ Kb,
    const unsigned short* __restrict__ Vt,
    unsigned short* __restrict__ AttOut) {     // [B,N,C] bf16
    __shared__ unsigned short Slds[16 * SEQ];  // 32 KB score strip
    __shared__ float rowmax[16];
    __shared__ float rowsum[16];

    const int lane = threadIdx.x & 31;
    const int g = lane >> 4;
    const int lr = lane & 15;
    const int rt = blockIdx.x;   // query tile
    const int bh = blockIdx.y;   // b*16+h
    const size_t headQK = (size_t)bh * SEQ * HEAD_DIM;
    const size_t headV  = (size_t)bh * HEAD_DIM * SEQ;

    // Q fragments for this 16-row tile (k = 0..31, 32..63)
    AFrag qa[2];
    {
        const unsigned short* Qp = Qb + headQK + (size_t)(rt * 16 + lr) * HEAD_DIM;
        qa[0].q[0] = *(const uint4*)(Qp + g * 16);
        qa[0].q[1] = *(const uint4*)(Qp + g * 16 + 8);
        qa[1].q[0] = *(const uint4*)(Qp + 32 + g * 16);
        qa[1].q[1] = *(const uint4*)(Qp + 32 + g * 16 + 8);
    }

    // ---- Pass 1: scores -> LDS (bf16), running row max ----
    float rmax[8];
#pragma unroll
    for (int e = 0; e < 8; ++e) rmax[e] = -3.0e38f;

    for (int mt = 0; mt < 64; ++mt) {
        const unsigned short* Kp = Kb + headQK + (size_t)(mt * 16 + lr) * HEAD_DIM;
        AFrag kb0, kb1;
        kb0.q[0] = *(const uint4*)(Kp + g * 16);
        kb0.q[1] = *(const uint4*)(Kp + g * 16 + 8);
        kb1.q[0] = *(const uint4*)(Kp + 32 + g * 16);
        kb1.q[1] = *(const uint4*)(Kp + 32 + g * 16 + 8);
        v8f s = v8f_zero();
        s = wmma_bf16(qa[0], kb0, s);
        s = wmma_bf16(qa[1], kb1, s);
#pragma unroll
        for (int e = 0; e < 8; ++e) {
            const float v = s[e] * ATT_SCALE;
            rmax[e] = fmaxf(rmax[e], v);
            Slds[(e + 8 * g) * SEQ + mt * 16 + lr] = f32_to_bf16(v);
        }
    }
    // reduce max over the 16 fragment columns (lanes with same g)
#pragma unroll
    for (int off = 1; off < 16; off <<= 1) {
#pragma unroll
        for (int e = 0; e < 8; ++e)
            rmax[e] = fmaxf(rmax[e], __shfl_xor(rmax[e], off, 32));
    }
    if (lr == 0) {
#pragma unroll
        for (int e = 0; e < 8; ++e) rowmax[e + 8 * g] = rmax[e];
    }
    __syncthreads();   // single-wave WG: hardware no-op, ordering via DScnt

    // ---- Pass 2: row sums of exp ----
    const float rm_row = rowmax[lr];    // this lane handles row lr, half g
    {
        const unsigned short* sp = &Slds[lr * SEQ + g * 512];
        float sum = 0.f;
        for (int it = 0; it < 64; ++it) {
            uint4 dv = *(const uint4*)(sp + it * 8);
            unsigned w0 = dv.x, w1 = dv.y, w2 = dv.z, w3 = dv.w;
            sum += __expf(bf16_to_f32((unsigned short)(w0 & 0xFFFF)) - rm_row);
            sum += __expf(bf16_to_f32((unsigned short)(w0 >> 16)) - rm_row);
            sum += __expf(bf16_to_f32((unsigned short)(w1 & 0xFFFF)) - rm_row);
            sum += __expf(bf16_to_f32((unsigned short)(w1 >> 16)) - rm_row);
            sum += __expf(bf16_to_f32((unsigned short)(w2 & 0xFFFF)) - rm_row);
            sum += __expf(bf16_to_f32((unsigned short)(w2 >> 16)) - rm_row);
            sum += __expf(bf16_to_f32((unsigned short)(w3 & 0xFFFF)) - rm_row);
            sum += __expf(bf16_to_f32((unsigned short)(w3 >> 16)) - rm_row);
        }
        sum += __shfl_xor(sum, 16, 32);
        if (lane < 16) rowsum[lane] = sum;
    }
    __syncthreads();

    // ---- Pass 3: O = P @ V ----
    v8f oacc[4];
#pragma unroll
    for (int dt = 0; dt < 4; ++dt) oacc[dt] = v8f_zero();

    for (int mc = 0; mc < 32; ++mc) {
        // Rebuild A fragment of P = exp(S - max) from LDS
        const unsigned short* prow = &Slds[lr * SEQ + mc * 32];
        uint4 a0 = *(const uint4*)(prow + g * 16);
        uint4 a1 = *(const uint4*)(prow + g * 16 + 8);
        unsigned wa[8] = {a0.x, a0.y, a0.z, a0.w, a1.x, a1.y, a1.z, a1.w};
        AFrag pa;
#pragma unroll
        for (int c = 0; c < 8; ++c) {
            const float lo = __expf(bf16_to_f32((unsigned short)(wa[c] & 0xFFFF)) - rm_row);
            const float hi = __expf(bf16_to_f32((unsigned short)(wa[c] >> 16)) - rm_row);
            pa.h[c * 2]     = f32_to_bf16(lo);
            pa.h[c * 2 + 1] = f32_to_bf16(hi);
        }
#pragma unroll
        for (int dt = 0; dt < 4; ++dt) {
            const unsigned short* Vp =
                Vt + headV + (size_t)(dt * 16 + lr) * SEQ + mc * 32 + g * 16;
            AFrag vb;
            vb.q[0] = *(const uint4*)(Vp);
            vb.q[1] = *(const uint4*)(Vp + 8);
            oacc[dt] = wmma_bf16(pa, vb, oacc[dt]);
        }
    }

    // ---- Epilogue: divide by row sum, store to AttOut [B,N,C] ----
    const int b = bh >> 4;
    const int h = bh & 15;
    float inv[8];
#pragma unroll
    for (int e = 0; e < 8; ++e) inv[e] = 1.0f / rowsum[e + 8 * g];
#pragma unroll
    for (int dt = 0; dt < 4; ++dt) {
#pragma unroll
        for (int e = 0; e < 8; ++e) {
            const int n = rt * 16 + 8 * g + e;
            const float val = oacc[dt][e] * inv[e];
            AttOut[((size_t)(b * SEQ + n)) * HIDDEN + h * HEAD_DIM + dt * 16 + lr] =
                f32_to_bf16(val);
        }
    }
}

// ---------------------------------------------------------------------------
// Kernel 3: out_proj  out = AttOut @ w_out^T + b_out  (f32 output)
// Same tiling as in_proj; A is bf16 already.
// ---------------------------------------------------------------------------
__global__ __launch_bounds__(256) void out_proj_kernel(
    const unsigned short* __restrict__ A,   // [8192,1024] bf16
    const unsigned short* __restrict__ W,   // [1024,1024] bf16
    const float* __restrict__ bias,         // [1024]
    float* __restrict__ out) {              // [8192,1024] f32
    const int lane = threadIdx.x & 31;
    const int wave = threadIdx.x >> 5;
    const int g = lane >> 4;
    const int lr = lane & 15;
    const int m0 = blockIdx.x * 128 + (wave & 1) * 64;
    const int n0 = blockIdx.y * 256 + (wave >> 1) * 64;

    v8f acc[4][4];
#pragma unroll
    for (int i = 0; i < 4; ++i)
#pragma unroll
        for (int j = 0; j < 4; ++j) acc[i][j] = v8f_zero();

    for (int k0 = 0; k0 < HIDDEN; k0 += 32) {
        AFrag a[4];
#pragma unroll
        for (int i = 0; i < 4; ++i) {
            const unsigned short* ap = A + (size_t)(m0 + i * 16 + lr) * HIDDEN + k0;
            a[i].q[0] = *(const uint4*)(ap + g * 16);
            a[i].q[1] = *(const uint4*)(ap + g * 16 + 8);
        }
        AFrag b[4];
#pragma unroll
        for (int j = 0; j < 4; ++j) {
            const unsigned short* bp =
                W + (size_t)(n0 + j * 16 + lr) * HIDDEN + k0 + g * 16;
            b[j].q[0] = *(const uint4*)(bp);
            b[j].q[1] = *(const uint4*)(bp + 8);
        }
#pragma unroll
        for (int i = 0; i < 4; ++i)
#pragma unroll
            for (int j = 0; j < 4; ++j)
                acc[i][j] = wmma_bf16(a[i], b[j], acc[i][j]);
    }

#pragma unroll
    for (int i = 0; i < 4; ++i) {
#pragma unroll
        for (int j = 0; j < 4; ++j) {
            const int o = n0 + j * 16 + lr;
            const float bv = bias[o];
#pragma unroll
            for (int e = 0; e < 8; ++e) {
                const int m = m0 + i * 16 + 8 * g + e;
                out[(size_t)m * HIDDEN + o] = acc[i][j][e] + bv;
            }
        }
    }
}

// ---------------------------------------------------------------------------
// Host launcher
// ---------------------------------------------------------------------------
extern "C" void kernel_launch(void* const* d_in, const int* in_sizes, int n_in,
                              void* d_out, int out_size, void* d_ws, size_t ws_size,
                              hipStream_t stream) {
    const float* x     = (const float*)d_in[0];   // [8,1024,1024]
    const float* w_in  = (const float*)d_in[1];   // [3072,1024]
    const float* b_in  = (const float*)d_in[2];   // [3072]
    const float* w_out = (const float*)d_in[3];   // [1024,1024]
    const float* b_out = (const float*)d_in[4];   // [1024]
    float* out = (float*)d_out;

    char* ws = (char*)d_ws;
    size_t off = 0;
    auto alloc = [&](size_t bytes) -> char* {
        char* p = ws + off;
        off += (bytes + 255) & ~(size_t)255;
        return p;
    };
    const size_t qkv_bytes = (size_t)BATCH * HEADS * SEQ * HEAD_DIM * 2; // 16 MiB
    unsigned short* Qb     = (unsigned short*)alloc(qkv_bytes);
    unsigned short* Kb     = (unsigned short*)alloc(qkv_bytes);
    unsigned short* Vt     = (unsigned short*)alloc(qkv_bytes);
    unsigned short* AttOut = (unsigned short*)alloc((size_t)M_TOTAL * HIDDEN * 2);
    unsigned short* Xb     = (unsigned short*)alloc((size_t)M_TOTAL * HIDDEN * 2);
    unsigned short* WbIn   = (unsigned short*)alloc((size_t)NPROJ * HIDDEN * 2);
    unsigned short* WbOut  = (unsigned short*)alloc((size_t)HIDDEN * HIDDEN * 2);
    if (off > ws_size) return;   // insufficient workspace

    // 0) casts: x, w_in, w_out -> bf16
    {
        int x4 = (M_TOTAL * HIDDEN) / 4;
        cast_f32_bf16_kernel<<<x4 / 256, 256, 0, stream>>>(x, Xb, x4);
        int n4 = (NPROJ * HIDDEN) / 4;
        cast_f32_bf16_kernel<<<n4 / 256, 256, 0, stream>>>(w_in, WbIn, n4);
        int m4 = (HIDDEN * HIDDEN) / 4;
        cast_f32_bf16_kernel<<<m4 / 256, 256, 0, stream>>>(w_out, WbOut, m4);
    }
    // 1) in projection -> Q / K / V^T (bf16)
    in_proj_kernel<<<dim3(M_TOTAL / 128, NPROJ / 256), 256, 0, stream>>>(
        Xb, WbIn, b_in, Qb, Kb, Vt);
    // 2) attention per (query-tile, head)
    attention_kernel<<<dim3(SEQ / 16, BATCH * HEADS), 32, 0, stream>>>(
        Qb, Kb, Vt, AttOut);
    // 3) out projection -> f32 output
    out_proj_kernel<<<dim3(M_TOTAL / 128, HIDDEN / 256), 256, 0, stream>>>(
        AttOut, WbOut, b_out, out);
}
